// MDGNN_21534966022956
// MI455X (gfx1250) — compile-verified, matrix operand order checked by reading, not run on hardware
//
#include <hip/hip_runtime.h>

// ---------------------------------------------------------------------------
// CDNA5 (gfx1250) WMMA bf16 GEMM + edge-softmax kernels for hetero-GNN forward
// ---------------------------------------------------------------------------

typedef __attribute__((ext_vector_type(16))) __bf16 v16bf;
typedef __attribute__((ext_vector_type(8)))  float  v8f;

__device__ __forceinline__ unsigned short f2bf(float f) {
  unsigned u = __float_as_uint(f);
  u += 0x7fffu + ((u >> 16) & 1u);          // round-to-nearest-even
  return (unsigned short)(u >> 16);
}

union BFrag {
  v16bf v;
  unsigned short s[16];
  uint4 q[2];
};

// C[M,N] = act( A[M,K] @ W[K,N] + bias (+ C if beta) )
// act: 0=none 1=relu 2=tanh.  K,N compile-time so accumulators stay in VGPRs.
template <int K, int N>
__global__ __launch_bounds__(256) void gemm_wmma_kernel(
    const float* __restrict__ A, const float* __restrict__ W,
    const float* __restrict__ bias, float* __restrict__ C,
    int M, int act, int beta)
{
  constexpr int KP  = (K + 31) & ~31;   // K padded to WMMA K-step
  constexpr int NTC = N / 16;           // #16-col tiles (<= 8)
  constexpr int KT  = KP / 32;          // #K-steps
  __shared__ __align__(16) unsigned short wlds[N * KP];  // W^T as bf16

  const int tid = threadIdx.x;
  // stage W transposed: wlds[col*KP + k], zero-pad k in [K,KP)
  for (int i = tid; i < N * KP; i += 256) {
    int col = i / KP;
    int k   = i - col * KP;
    wlds[i] = (k < K) ? f2bf(W[(size_t)k * N + col]) : (unsigned short)0;
  }
  __syncthreads();

  const int wid   = tid >> 5;
  const int lane  = tid & 31;
  const int half  = lane >> 4;      // A/B fragment K-half select
  const int l15   = lane & 15;
  const int mbase = blockIdx.x * 128 + wid * 16;
  const int arow  = mbase + l15;

  v8f acc[NTC];
#pragma unroll
  for (int i = 0; i < NTC; ++i)
#pragma unroll
    for (int j = 0; j < 8; ++j) acc[i][j] = 0.f;

#pragma unroll
  for (int kt = 0; kt < KT; ++kt) {
    // ---- A fragment: lanes 0-15 K {0..7,16..23}; lanes 16-31 K {8..15,24..31}
    BFrag a;
    const int kb = kt * 32 + half * 8;
    if constexpr (K % 32 == 0) {
      if (arow < M) {
        const float4* ap  = (const float4*)(A + (size_t)arow * K + kb);
        const float4* ap2 = (const float4*)(A + (size_t)arow * K + kb + 16);
        float4 f0 = ap[0], f1 = ap[1], f2 = ap2[0], f3 = ap2[1];
        a.s[0]  = f2bf(f0.x); a.s[1]  = f2bf(f0.y); a.s[2]  = f2bf(f0.z); a.s[3]  = f2bf(f0.w);
        a.s[4]  = f2bf(f1.x); a.s[5]  = f2bf(f1.y); a.s[6]  = f2bf(f1.z); a.s[7]  = f2bf(f1.w);
        a.s[8]  = f2bf(f2.x); a.s[9]  = f2bf(f2.y); a.s[10] = f2bf(f2.z); a.s[11] = f2bf(f2.w);
        a.s[12] = f2bf(f3.x); a.s[13] = f2bf(f3.y); a.s[14] = f2bf(f3.z); a.s[15] = f2bf(f3.w);
      } else {
#pragma unroll
        for (int j = 0; j < 16; ++j) a.s[j] = 0;
      }
    } else {
#pragma unroll
      for (int j = 0; j < 8; ++j) {
        const int k0 = kb + j, k1 = kb + 16 + j;
        a.s[j]     = (arow < M && k0 < K) ? f2bf(A[(size_t)arow * K + k0]) : (unsigned short)0;
        a.s[8 + j] = (arow < M && k1 < K) ? f2bf(A[(size_t)arow * K + k1]) : (unsigned short)0;
      }
    }
    // ---- per N-tile: B fragment from LDS (two ds_load_b128) + WMMA
#pragma unroll
    for (int nt = 0; nt < NTC; ++nt) {
      BFrag b;
      const int col = nt * 16 + l15;
      const unsigned short* bp = &wlds[col * KP + kt * 32 + half * 8];
      b.q[0] = *(const uint4*)bp;
      b.q[1] = *(const uint4*)(bp + 16);
      acc[nt] = __builtin_amdgcn_wmma_f32_16x16x32_bf16(
          false, a.v, false, b.v, (short)0, acc[nt], false, false);
    }
  }

  // ---- writeback: VGPR r -> row (half?8+r:r), lane&15 -> col
#pragma unroll
  for (int nt = 0; nt < NTC; ++nt) {
    const int col = nt * 16 + l15;
    const float bv = bias ? bias[col] : 0.f;
#pragma unroll
    for (int r = 0; r < 8; ++r) {
      const int mrow = mbase + half * 8 + r;
      if (mrow < M) {
        float v = acc[nt][r] + bv;
        if (beta) v += C[(size_t)mrow * N + col];
        if (act == 1) v = v > 0.f ? v : 0.f;
        else if (act == 2) v = tanhf(v);
        C[(size_t)mrow * N + col] = v;
      }
    }
  }
}

__global__ void fill_kernel(float* __restrict__ p, float v, int n) {
  int t = blockIdx.x * blockDim.x + threadIdx.x;
  if (t < n) p[t] = v;
}

__device__ __forceinline__ void atomicMaxF(float* addr, float val) {
  if (val >= 0.f) atomicMax((int*)addr, __float_as_int(val));
  else            atomicMin((unsigned int*)addr, __float_as_uint(val));
}

// score[e,h] = sum_d tanh(sf[si,h,d])*aw[d] + tanh(df[di,h,d])*aw[32+d] + ab
__global__ void edge_score_kernel(
    const float* __restrict__ sf, const float* __restrict__ df,
    const int* __restrict__ si, const int* __restrict__ di,
    const float* __restrict__ aw, const float* __restrict__ ab,
    float* __restrict__ score, float* __restrict__ mx, int E)
{
  int t = blockIdx.x * blockDim.x + threadIdx.x;
  if (t >= E * 4) return;
  int e = t >> 2, h = t & 3;
  int s = si[e], d = di[e];
  const float* sp = sf + (size_t)s * 128 + h * 32;
  const float* dp = df + (size_t)d * 128 + h * 32;
  float acc = ab[0];
#pragma unroll 8
  for (int i = 0; i < 32; ++i)
    acc += tanhf(sp[i]) * aw[i] + tanhf(dp[i]) * aw[32 + i];
  score[t] = acc;
  atomicMaxF(&mx[d * 4 + h], acc);
}

__global__ void edge_exp_kernel(
    const float* __restrict__ score, const float* __restrict__ mx,
    const int* __restrict__ di, float* __restrict__ ex,
    float* __restrict__ den, int E)
{
  int t = blockIdx.x * blockDim.x + threadIdx.x;
  if (t >= E * 4) return;
  int e = t >> 2, h = t & 3;
  int d = di[e];
  float v = __expf(score[t] - mx[d * 4 + h]);
  ex[t] = v;
  atomicAdd(&den[d * 4 + h], v);
}

__global__ void edge_agg_kernel(
    const float* __restrict__ ex, const float* __restrict__ den,
    const float* __restrict__ msg, const int* __restrict__ si,
    const int* __restrict__ di, float* __restrict__ agg, int E)
{
  int t = blockIdx.x * blockDim.x + threadIdx.x;
  if (t >= E * 4) return;
  int e = t >> 2, h = t & 3;
  int s = si[e], d = di[e];
  float alpha = ex[t] / den[d * 4 + h];
  const float* mp = msg + (size_t)s * 128 + h * 32;
  float* ap = agg + (size_t)d * 128 + h * 32;
#pragma unroll 8
  for (int i = 0; i < 32; ++i) atomicAdd(&ap[i], mp[i] * alpha);
}

__global__ void head_mean_kernel(const float* __restrict__ agg,
                                 float* __restrict__ out, int n)
{
  int t = blockIdx.x * blockDim.x + threadIdx.x;
  if (t >= n * 32) return;
  int node = t >> 5, f = t & 31;
  const float* a = agg + (size_t)node * 128 + f;
  out[t] = 0.25f * (a[0] + a[32] + a[64] + a[96]);
}

// out[row] = LN(x[row] (+ res[row])) * g + b ; D=128, one wave32 per row
__global__ __launch_bounds__(256) void ln_kernel(
    const float* __restrict__ x, const float* __restrict__ res,
    const float* __restrict__ g, const float* __restrict__ b,
    float* __restrict__ out, int n)
{
  int wid = threadIdx.x >> 5, lane = threadIdx.x & 31;
  int row = blockIdx.x * 8 + wid;
  if (row >= n) return;
  float4 v = ((const float4*)(x + (size_t)row * 128))[lane];
  if (res) {
    float4 r = ((const float4*)(res + (size_t)row * 128))[lane];
    v.x += r.x; v.y += r.y; v.z += r.z; v.w += r.w;
  }
  float s = v.x + v.y + v.z + v.w;
  for (int o = 16; o; o >>= 1) s += __shfl_xor(s, o, 32);
  float mean = s * (1.f / 128.f);
  float dx = v.x - mean, dy = v.y - mean, dz = v.z - mean, dw = v.w - mean;
  float q = dx * dx + dy * dy + dz * dz + dw * dw;
  for (int o = 16; o; o >>= 1) q += __shfl_xor(q, o, 32);
  float inv = rsqrtf(q * (1.f / 128.f) + 1e-5f);
  float4 gv = ((const float4*)g)[lane];
  float4 bv = ((const float4*)b)[lane];
  float4 o4;
  o4.x = dx * inv * gv.x + bv.x;
  o4.y = dy * inv * gv.y + bv.y;
  o4.z = dz * inv * gv.z + bv.z;
  o4.w = dw * inv * gv.w + bv.w;
  ((float4*)(out + (size_t)row * 128))[lane] = o4;
}

// hs[n*4+p] = h[n,:] . w   (h = tanh-projected path)
__global__ void rowdot_kernel(const float* __restrict__ h,
                              const float* __restrict__ w,
                              float* __restrict__ hs, int n, int p)
{
  int t = blockIdx.x * blockDim.x + threadIdx.x;
  if (t >= n) return;
  const float4* hp = (const float4*)(h + (size_t)t * 128);
  const float4* wp = (const float4*)w;
  float acc = 0.f;
#pragma unroll 8
  for (int i = 0; i < 32; ++i) {
    float4 a = hp[i], b = wp[i];
    acc += a.x * b.x + a.y * b.y + a.z * b.z + a.w * b.w;
  }
  hs[t * 4 + p] = acc;
}

// comb[n,f] = sum_p softmax_p(hs[n,:]) * path_p[n,f]
__global__ void meta_combine_kernel(
    const float* __restrict__ hs,
    const float* __restrict__ p0, const float* __restrict__ p1,
    const float* __restrict__ p2, const float* __restrict__ p3,
    float* __restrict__ comb, int n)
{
  int t = blockIdx.x * blockDim.x + threadIdx.x;
  if (t >= n * 128) return;
  int node = t >> 7;
  float s0 = hs[node * 4 + 0], s1 = hs[node * 4 + 1];
  float s2 = hs[node * 4 + 2], s3 = hs[node * 4 + 3];
  float m = fmaxf(fmaxf(s0, s1), fmaxf(s2, s3));
  float e0 = __expf(s0 - m), e1 = __expf(s1 - m);
  float e2 = __expf(s2 - m), e3 = __expf(s3 - m);
  float inv = 1.f / (e0 + e1 + e2 + e3);
  size_t idx = (size_t)t;
  comb[idx] = (e0 * p0[idx] + e1 * p1[idx] + e2 * p2[idx] + e3 * p3[idx]) * inv;
}

// ---------------------------------------------------------------------------
// Host orchestration
// ---------------------------------------------------------------------------
static inline int cdiv(int a, int b) { return (a + b - 1) / b; }

static void gemm(hipStream_t st, const float* A, const float* W, const float* bias,
                 float* C, int M, int K, int N, int act, int beta) {
  dim3 g(cdiv(M, 128)), blk(256);
  if (K == 128 && N == 128)
    gemm_wmma_kernel<128, 128><<<g, blk, 0, st>>>(A, W, bias, C, M, act, beta);
  else if (K == 32 && N == 128)
    gemm_wmma_kernel<32, 128><<<g, blk, 0, st>>>(A, W, bias, C, M, act, beta);
  else if (K == 16 && N == 128)
    gemm_wmma_kernel<16, 128><<<g, blk, 0, st>>>(A, W, bias, C, M, act, beta);
  else if (K == 32 && N == 32)
    gemm_wmma_kernel<32, 32><<<g, blk, 0, st>>>(A, W, bias, C, M, act, beta);
}
static void fillb(hipStream_t st, float* p, float v, int n) {
  fill_kernel<<<cdiv(n, 256), 256, 0, st>>>(p, v, n);
}

struct RGAP {  // pointers in JAX pytree (sorted-key) order
  const float *attn_b, *attn_w, *dst_b, *dst_w, *ln_b, *ln_g,
              *merge_b, *merge_w, *msg_b, *msg_w, *out_b, *out_w,
              *src_b, *src_w;
};
struct METAP { const float *ln_b, *ln_g, *out_b, *out_w, *path_b, *path_w, *score_w; };
struct ENCP  { const float *b1, *b2, *w1, *w2; };

struct Scratch {
  float *T0, *T1, *T2;            // [NS,128] each (T1 doubles as agg)
  float *MX, *DEN, *AGGM, *SC, *EX;
};

static void rga(hipStream_t st, const float* XS, int Ns, const float* XD, int Nd,
                const int* ei, int E, const RGAP& P, float* OUT, const Scratch& W)
{
  const int* si = ei;
  const int* di = ei + E;
  gemm(st, XS, P.src_w, P.src_b, W.T0, Ns, 128, 128, 0, 0);        // sf
  gemm(st, XD, P.dst_w, P.dst_b, W.T1, Nd, 128, 128, 0, 0);        // df
  gemm(st, W.T0, P.msg_w, P.msg_b, W.T2, Ns * 4, 32, 32, 0, 0);    // msg (per-head)
  fillb(st, W.MX, -1e30f, Nd * 4);
  fillb(st, W.DEN, 0.f, Nd * 4);
  int T = E * 4;
  edge_score_kernel<<<cdiv(T, 256), 256, 0, st>>>(W.T0, W.T1, si, di,
                                                  P.attn_w, P.attn_b, W.SC, W.MX, E);
  edge_exp_kernel<<<cdiv(T, 256), 256, 0, st>>>(W.SC, W.MX, di, W.EX, W.DEN, E);
  fillb(st, W.T1, 0.f, Nd * 128);                                  // df dead -> agg
  edge_agg_kernel<<<cdiv(T, 256), 256, 0, st>>>(W.EX, W.DEN, W.T2, si, di, W.T1, E);
  head_mean_kernel<<<cdiv(Nd * 32, 256), 256, 0, st>>>(W.T1, W.AGGM, Nd);
  gemm(st, W.AGGM, P.merge_w, P.merge_b, W.T0, Nd, 32, 128, 0, 0); // sf dead -> merged
  gemm(st, XD, P.out_w, P.out_b, W.T2, Nd, 128, 128, 0, 0);        // msg dead -> upd
  gemm(st, W.T0, P.out_w + 128 * 128, nullptr, W.T2, Nd, 128, 128, 0, 1);
  ln_kernel<<<cdiv(Nd, 8), 256, 0, st>>>(W.T2, XD, P.ln_g, P.ln_b, OUT, Nd);
}

static void meta(hipStream_t st, const float* P0, const float* P1, const float* P2,
                 const float* P3, int n, const METAP& M, float* OUT, const Scratch& W)
{
  const float* paths[4] = {P0, P1, P2, P3};
  for (int p = 0; p < 4; ++p) {
    gemm(st, paths[p], M.path_w, M.path_b, W.T0, n, 128, 128, 2, 0);  // tanh
    rowdot_kernel<<<cdiv(n, 256), 256, 0, st>>>(W.T0, M.score_w, W.MX, n, p);
  }
  meta_combine_kernel<<<cdiv(n * 128, 256), 256, 0, st>>>(W.MX, P0, P1, P2, P3, W.T1, n);
  gemm(st, W.T1, M.out_w, M.out_b, W.T0, n, 128, 128, 0, 0);
  ln_kernel<<<cdiv(n, 8), 256, 0, st>>>(W.T0, nullptr, M.ln_g, M.ln_b, OUT, n);
}

extern "C" void kernel_launch(void* const* d_in, const int* in_sizes, int n_in,
                              void* d_out, int out_size, void* d_ws, size_t ws_size,
                              hipStream_t stream)
{
  const int NS = 50000, NB = 2000, NI = 300;

  const float* stock = (const float*)d_in[0];
  const float* bank  = (const float*)d_in[1];
  const float* indus = (const float*)d_in[2];
  const int* ei[6];
  int E[6];
  for (int i = 0; i < 6; ++i) {
    ei[i] = (const int*)d_in[3 + i];
    E[i]  = in_sizes[3 + i] / 2;
  }
  // order: ei_ss, ei_sb, ei_si, ei_bs, ei_is, ei_ii
  enum { ESS = 0, ESB = 1, ESI = 2, EBS = 3, EIS = 4, EII = 5 };

  // ---- params: pytree flatten order (dict keys sorted, lists in order) ----
  int pidx = 9;
  auto nx = [&]() { return (const float*)d_in[pidx++]; };
  ENCP enc_b{nx(), nx(), nx(), nx()};   // b1,b2,w1,w2
  ENCP enc_i{nx(), nx(), nx(), nx()};
  ENCP enc_s{nx(), nx(), nx(), nx()};
  RGAP L[2][6];                          // per layer: BS, II, IS, SB, SI, SS
  for (int l = 0; l < 2; ++l)
    for (int t = 0; t < 6; ++t) {
      RGAP& r = L[l][t];
      r.attn_b = nx(); r.attn_w = nx(); r.dst_b = nx(); r.dst_w = nx();
      r.ln_b = nx();   r.ln_g = nx();   r.merge_b = nx(); r.merge_w = nx();
      r.msg_b = nx();  r.msg_w = nx();  r.out_b = nx();  r.out_w = nx();
      r.src_b = nx();  r.src_w = nx();
    }
  enum { kBS = 0, kII = 1, kIS = 2, kSB = 3, kSI = 4, kSS = 5 };
  METAP MP{nx(), nx(), nx(), nx(), nx(), nx(), nx()};

  // ---- scratch carve ----
  size_t off = 0;
  auto carve = [&](size_t elems) -> float* {
    float* p = (float*)((char*)d_ws + off);
    off += ((elems * sizeof(float) + 255) & ~(size_t)255);
    return p;
  };
  const size_t ND = (size_t)NS * 128;
  float* S0 = carve(ND);                 // sh
  float* S1 = carve(ND);                 // path SS
  float* S2 = carve(ND);                 // path SB
  float* S3 = carve(ND);                 // path SI
  Scratch W;
  W.T0 = carve(ND); W.T1 = carve(ND); W.T2 = carve(ND);
  float* BH0 = carve((size_t)NB * 128);
  float* BH1 = carve((size_t)NB * 128);
  float* IH0 = carve((size_t)NI * 128);
  float* IH1 = carve((size_t)NI * 128);
  W.MX   = carve((size_t)NS * 4);
  W.DEN  = carve((size_t)NS * 4);
  W.AGGM = carve((size_t)NS * 32);
  int EMAX = E[0];
  for (int i = 1; i < 6; ++i) if (E[i] > EMAX) EMAX = E[i];
  W.SC = carve((size_t)EMAX * 4);
  W.EX = carve((size_t)EMAX * 4);

  // ---- encoders: relu(x@w1+b1)@w2+b2 ----
  gemm(stream, stock, enc_s.w1, enc_s.b1, W.T0, NS, 32, 128, 1, 0);
  gemm(stream, W.T0, enc_s.w2, enc_s.b2, S0, NS, 128, 128, 0, 0);
  gemm(stream, bank, enc_b.w1, enc_b.b1, W.T0, NB, 16, 128, 1, 0);
  gemm(stream, W.T0, enc_b.w2, enc_b.b2, BH0, NB, 128, 128, 0, 0);
  gemm(stream, indus, enc_i.w1, enc_i.b1, W.T0, NI, 16, 128, 1, 0);
  gemm(stream, W.T0, enc_i.w2, enc_i.b2, IH0, NI, 128, 128, 0, 0);

  float* sh = S0;
  float* bh = BH0; float* bh_alt = BH1;
  float* ih = IH0; float* ih_alt = IH1;

  for (int l = 0; l < 2; ++l) {
    rga(stream, sh, NS, sh, NS, ei[ESS], E[ESS], L[l][kSS], S1, W);
    rga(stream, bh, NB, sh, NS, ei[ESB], E[ESB], L[l][kSB], S2, W);
    rga(stream, ih, NI, sh, NS, ei[ESI], E[ESI], L[l][kSI], S3, W);
    float* newsh = (l == 1) ? (float*)d_out : S0;
    meta(stream, sh, S1, S2, S3, NS, MP, newsh, W);
    sh = newsh;
    if (l == 0) {  // last-layer bank/industry updates don't affect the output
      rga(stream, sh, NS, bh, NB, ei[EBS], E[EBS], L[l][kBS], bh_alt, W);
      { float* t = bh; bh = bh_alt; bh_alt = t; }
      rga(stream, sh, NS, ih, NI, ei[EIS], E[EIS], L[l][kIS], ih_alt, W);
      { float* t = ih; ih = ih_alt; ih_alt = t; }
      rga(stream, ih, NI, ih, NI, ei[EII], E[EII], L[l][kII], ih_alt, W);
      { float* t = ih; ih = ih_alt; ih_alt = t; }
    }
  }
  (void)n_in; (void)out_size; (void)ws_size;
}
